// Decoder_29832842838551
// MI455X (gfx1250) — matrix operand order repaired and checked by reading, not compiled, not run on hardware
//
#include <hip/hip_runtime.h>

// ---------------------------------------------------------------------------
// Fully-fused decoder for MI455X (gfx1250, wave32, WMMA).
//   fc -> xproj -> 256-step LSTM (WMMA f16 16x16x32, K=H=32 exactly)
//      -> conv1d(32->16,k5,p2)+relu (5 chained WMMAs over a register A-ring)
//      -> conv1d(16->1,k5,p2) -> out [8192,1,256]
// One wave owns 16 batch rows; everything stays in VGPRs/LDS, zero workspace.
// ---------------------------------------------------------------------------

typedef __attribute__((ext_vector_type(16))) _Float16 v16h;
typedef __attribute__((ext_vector_type(8)))  float    v8f;

#define HH    32
#define WW    256
#define BATCH 8192
#define NW    2          // waves per block
#define HP    36         // padded h-slab row stride (floats), even & mult-of-4
#define Y1P   17         // padded y1 ring stride over batch dim

__device__ __forceinline__ float fast_sigmoid(float x) {
    // 1/(1+exp(-x)) ; exp via v_exp_f32 (2^x). Saturates cleanly at +-inf.
    return __builtin_amdgcn_rcpf(1.0f + __builtin_amdgcn_exp2f(-1.4426950408889634f * x));
}
__device__ __forceinline__ float fast_tanh(float x) {
    // 1 - 2/(exp(2x)+1). exp2 overflow -> rcp(inf)=0 -> 1; underflow -> -1.
    return 1.0f - 2.0f * __builtin_amdgcn_rcpf(1.0f + __builtin_amdgcn_exp2f(2.8853900817779268f * x));
}
__device__ __forceinline__ v8f wmma16(v16h a, v16h b, v8f c) {
    return __builtin_amdgcn_wmma_f32_16x16x32_f16(false, a, false, b, (short)0, c, false, false);
}
__device__ __forceinline__ v16h zero16() {
    v16h a;
#pragma unroll
    for (int e = 0; e < 16; ++e) a[e] = (_Float16)0.0f;
    return a;
}

__global__ __launch_bounds__(NW * 32)
void decoder_fused(const float* __restrict__ z,    const float* __restrict__ fc_w,
                   const float* __restrict__ fc_b, const float* __restrict__ w_ih,
                   const float* __restrict__ w_hh, const float* __restrict__ b_ih,
                   const float* __restrict__ b_hh, const float* __restrict__ c1w,
                   const float* __restrict__ c1b,  const float* __restrict__ c2w,
                   const float* __restrict__ c2b,  float* __restrict__ out)
{
    __shared__ float s_h [NW][16 * HP];        // current-step h (C-layout -> A-layout transpose)
    __shared__ float s_y1[NW][5 * 16 * Y1P];   // conv1 output ring: [slot][ich][batch(padded)]
    __shared__ float s_xp[NW][2048];           // overlay: xs[16][65] (prologue), then xproj [8][8][32]
    __shared__ float s_w2[80];

    const int tid   = threadIdx.x;
    const int wv    = tid >> 5;
    const int lane  = tid & 31;
    const int n     = lane & 15;   // WMMA N-index / A-matrix M row
    const int hf    = lane >> 4;   // lane half
    const int tile  = blockIdx.x * NW + wv;
    const int bbase = tile * 16;

    float* xs  = s_xp[wv];
    float* xpl = s_xp[wv];
    float* hsl = s_h[wv];
    float* y1s = s_y1[wv];

    // ---- stage conv2 weights in LDS (broadcast reads later) ----
    for (int i = tid; i < 80; i += NW * 32) s_w2[i] = c2w[i];
    // ---- zero y1 ring (provides conv2 zero-padding for t<0) ----
    for (int i = lane; i < 5 * 16 * Y1P; i += 32) y1s[i] = 0.0f;

    // ---- x = z @ fc_w.T + fc_b  -> xs[16][65] (padded) ----
    for (int idx = lane; idx < 16 * 64; idx += 32) {
        const int row = idx >> 6, col = idx & 63;
        float acc = fc_b[col];
#pragma unroll
        for (int k = 0; k < 16; ++k)
            acc += z[(bbase + row) * 16 + k] * fc_w[col * 16 + k];
        xs[row * 65 + col] = acc;
    }
    __syncthreads();

    // ---- xproj = x @ w_ih.T + b_ih + b_hh, in C-tile layout (temp regs) ----
    float xpt[8][8];
#pragma unroll
    for (int nt = 0; nt < 8; ++nt)
#pragma unroll
        for (int r = 0; r < 8; ++r) {
            const int col = nt * 16 + n;
            const int M   = r + 8 * hf;
            float acc = b_ih[col] + b_hh[col];
            for (int k = 0; k < 64; ++k)
                acc += xs[M * 65 + k] * w_ih[col * 64 + k];
            xpt[nt][r] = acc;
        }
    __syncthreads();   // xs fully consumed; overlay becomes xproj storage
#pragma unroll
    for (int nt = 0; nt < 8; ++nt)
#pragma unroll
        for (int r = 0; r < 8; ++r)
            xpl[(nt * 8 + r) * 32 + lane] = xpt[nt][r];

    // ---- constant B tiles: w_hh^T (gate cols), lane = N, elem = K(channel, table order) ----
    v16h bt[8];
#pragma unroll
    for (int nt = 0; nt < 8; ++nt)
#pragma unroll
        for (int e = 0; e < 16; ++e)
            bt[nt][e] = (_Float16)w_hh[(nt * 16 + n) * HH + ((e >> 3) << 4) + 8 * hf + (e & 7)];

    // ---- constant conv1 B tiles per tap: B[K=c][N=och] = c1w[och][c][dt] ----
    v16h bc[5];
#pragma unroll
    for (int dt = 0; dt < 5; ++dt)
#pragma unroll
        for (int e = 0; e < 16; ++e)
            bc[dt][e] = (_Float16)c1w[(n * HH + ((e >> 3) << 4) + 8 * hf + (e & 7)) * 5 + dt];

    float cst[2][8];
#pragma unroll
    for (int cf = 0; cf < 2; ++cf)
#pragma unroll
        for (int r = 0; r < 8; ++r) cst[cf][r] = 0.0f;

    // A-matrix register shift-ring: a0..a4 hold h[t-4..t] (f16 A-layout)
    v16h a0 = zero16(), a1 = zero16(), a2 = zero16(), a3 = zero16(), a4 = zero16();

    const float b1v = c1b[n];
    const float b2v = c2b[0];
    __syncthreads();

    // ======================= fused streaming loop =======================
    for (int t = 0; t < WW + 4; ++t) {
        __syncthreads();   // WAR: h-slab + oldest y1 slot reuse

        // ---------------- LSTM step: gates = h@w_hh^T + xproj ----------------
        if (t < WW) {
#pragma unroll
            for (int cf = 0; cf < 2; ++cf) {     // channel halves 0-15 / 16-31
                v8f xi, xf, xg, xo;
#pragma unroll
                for (int r = 0; r < 8; ++r) {
                    xi[r] = xpl[((0 + cf) * 8 + r) * 32 + lane];
                    xf[r] = xpl[((2 + cf) * 8 + r) * 32 + lane];
                    xg[r] = xpl[((4 + cf) * 8 + r) * 32 + lane];
                    xo[r] = xpl[((6 + cf) * 8 + r) * 32 + lane];
                }
                v8f gi = wmma16(a4, bt[0 + cf], xi);
                v8f gf = wmma16(a4, bt[2 + cf], xf);
                v8f gg = wmma16(a4, bt[4 + cf], xg);
                v8f go = wmma16(a4, bt[6 + cf], xo);
#pragma unroll
                for (int r = 0; r < 8; ++r) {
                    const float iv = fast_sigmoid(gi[r]);
                    const float fv = fast_sigmoid(gf[r]);
                    const float gv = fast_tanh(gg[r]);
                    const float ov = fast_sigmoid(go[r]);
                    const float cv = fv * cst[cf][r] + iv * gv;
                    cst[cf][r] = cv;
                    // C-layout: value (r,lane) -> h[batch=r+8*hf][ch=cf*16+n]
                    hsl[(r + 8 * hf) * HP + cf * 16 + n] = ov * fast_tanh(cv);
                }
            }
        }
        __syncthreads();   // h-slab writes visible across lanes

        // -------- shift A ring; rebuild A-layout f16 tile from h[t] --------
        a0 = a1; a1 = a2; a2 = a3; a3 = a4;
        if (t < WW) {
            v16h a;
            const float* hrow = hsl + n * HP;    // A: row M = lane&15
#pragma unroll
            for (int v = 0; v < 8; ++v) {
                const int K0 = ((v >> 2) << 4) + 8 * hf + ((v & 3) << 1);
                const float2 p = *(const float2*)(hrow + K0);
                a[2 * v]     = (_Float16)p.x;
                a[2 * v + 1] = (_Float16)p.y;
            }
            a4 = a;
        } else {
            a4 = zero16();   // h[t>=W] = 0 (conv zero padding)
        }

        // -------- conv1 (+bias,+relu) at time t-2 via 5 chained WMMAs --------
        if (t >= 2) {
            v8f acc;
            const int q = (t - 2) % 5;
            if (t < WW + 2) {
#pragma unroll
                for (int r = 0; r < 8; ++r) acc[r] = b1v;
                acc = wmma16(a0, bc[0], acc);
                acc = wmma16(a1, bc[1], acc);
                acc = wmma16(a2, bc[2], acc);
                acc = wmma16(a3, bc[3], acc);
                acc = wmma16(a4, bc[4], acc);
#pragma unroll
                for (int r = 0; r < 8; ++r) acc[r] = fmaxf(acc[r], 0.0f);
            } else {
#pragma unroll
                for (int r = 0; r < 8; ++r) acc[r] = 0.0f;   // y1[t>=W] = 0
            }
#pragma unroll
            for (int r = 0; r < 8; ++r)   // [slot][ich=n][batch=r+8*hf]
                y1s[(q * 16 + n) * Y1P + r + 8 * hf] = acc[r];
        }
        __syncthreads();   // y1 slot visible

        // ---------------- conv2 -> out[b][t-4] ----------------
        if (t >= 4 && lane < 16) {
            const int tt = t - 4;
            const int b  = lane;
            float acc = b2v;
#pragma unroll
            for (int dt = 0; dt < 5; ++dt) {
                const int s = (tt - 2 + dt + 10) % 5;
#pragma unroll
                for (int ic = 0; ic < 16; ++ic)
                    acc += y1s[(s * 16 + ic) * Y1P + b] * s_w2[ic * 5 + dt];
            }
            out[(bbase + b) * WW + tt] = acc;
        }
    }
}

extern "C" void kernel_launch(void* const* d_in, const int* in_sizes, int n_in,
                              void* d_out, int out_size, void* d_ws, size_t ws_size,
                              hipStream_t stream) {
    (void)in_sizes; (void)n_in; (void)out_size; (void)d_ws; (void)ws_size;
    const float* z    = (const float*)d_in[0];
    const float* fc_w = (const float*)d_in[1];
    const float* fc_b = (const float*)d_in[2];
    const float* w_ih = (const float*)d_in[3];
    const float* w_hh = (const float*)d_in[4];
    const float* b_ih = (const float*)d_in[5];
    const float* b_hh = (const float*)d_in[6];
    const float* c1w  = (const float*)d_in[7];
    const float* c1b  = (const float*)d_in[8];
    const float* c2w  = (const float*)d_in[9];
    const float* c2b  = (const float*)d_in[10];
    float* outp = (float*)d_out;

    const int blocks = BATCH / 16 / NW;   // 256 blocks x 64 threads = 512 waves
    decoder_fused<<<blocks, NW * 32, 0, stream>>>(z, fc_w, fc_b, w_ih, w_hh,
                                                  b_ih, b_hh, c1w, c1b, c2w, c2b, outp);
}
